// KeyedAttention_30605936951433
// MI455X (gfx1250) — compile-verified, matrix operand order checked by reading, not run on hardware
//
#include <hip/hip_runtime.h>
#include <hip/hip_bf16.h>

// Problem constants (match reference)
#define N_INST 6144
#define P_DIM  1024
#define D_DIM  128
#define B_BAGS 512
#define EPS    1e-6f
#define BN_EPS 1e-5f
#define INV_BATCH (1.0f / 512.0f)

typedef _Float16 v16h __attribute__((ext_vector_type(16)));
typedef float    v8f  __attribute__((ext_vector_type(8)));

__device__ __forceinline__ v8f wmma_f16(v16h a, v16h b, v8f c) {
    // D = A(16x32 f16) x B(32x16 f16) + C(16x16 f32)
    return __builtin_amdgcn_wmma_f32_16x16x32_f16(
        /*neg_a=*/false, a, /*neg_b=*/false, b,
        /*c_mod=*/(short)0, c, /*reuse_a=*/false, /*reuse_b=*/false);
}

// CDNA5 async copy: memory -> LDS without a VGPR round-trip (ASYNCcnt).
// Per-lane 16B transfer; lds_dst/gsrc must be 16B aligned.
__device__ __forceinline__ void async_ld16(const void* lds_dst, const void* gsrc) {
    const unsigned lds = (unsigned)(size_t)lds_dst;           // low 32 bits = LDS offset
    const unsigned long long g = (unsigned long long)(size_t)gsrc;
    asm volatile("global_load_async_to_lds_b128 %0, %1, off"
                 :: "v"(lds), "v"(g) : "memory");
}

__device__ __forceinline__ void wait_async0() {
    asm volatile("s_wait_asynccnt 0x0" ::: "memory");
}

// Build a 16x32 f16 fragment from a row-major LDS tile [64][40] (padded rows).
// ISA 16-bit A layout: lanes 0-15 -> row=lane, K in {0..7,16..23};
// lanes 16-31 -> row=lane-16, K in {8..15,24..31}. B fragment mirrors with
// "row" = output column. Both load as two contiguous 16B chunks.
__device__ __forceinline__ v16h frag_ld(const _Float16 (*buf)[40], int row, int kb) {
    v16h v;
    *((uint4*)&v)       = *((const uint4*)&buf[row][kb]);
    *(((uint4*)&v) + 1) = *((const uint4*)&buf[row][kb + 16]);
    return v;
}

// ---------------------------------------------------------------------------
// Generic C(M,N) = A(M,K) x B(N,K)^T (+ bias[col]), f16 in / f32 out.
// grid = (Ncols/64, Mrows/64), block = 128 (4 waves). Dims multiples of 64/32.
// Double-buffered LDS staging via async-to-LDS; one barrier per k-step.
// ---------------------------------------------------------------------------
__global__ __launch_bounds__(128)
void ka_gemm_nt(const _Float16* __restrict__ A, const _Float16* __restrict__ B,
                const float* __restrict__ bias, float* __restrict__ C,
                int K, int lda, int ldb, int ldc) {
    __shared__ _Float16 As[2][64][40];
    __shared__ _Float16 Bs[2][64][40];
    const int tid  = threadIdx.x;
    const int lane = tid & 31;
    const int wave = tid >> 5;
    const int fr   = lane & 15;
    const int kb   = (lane >> 4) << 3;
    const int i0 = blockIdx.y * 64, j0 = blockIdx.x * 64;
    const int srow = tid >> 1, scol = (tid & 1) << 4;

    const _Float16* arow = A + (size_t)(i0 + srow) * lda + scol;
    const _Float16* brow = B + (size_t)(j0 + srow) * ldb + scol;

    v8f acc[4];
    #pragma unroll
    for (int c = 0; c < 4; ++c)
        #pragma unroll
        for (int e = 0; e < 8; ++e) acc[c][e] = 0.f;

    const int nsteps = K >> 5;
    // prologue: stage k-step 0 into buffer 0
    async_ld16(&As[0][srow][scol],     arow);
    async_ld16(&As[0][srow][scol + 8], arow + 8);
    async_ld16(&Bs[0][srow][scol],     brow);
    async_ld16(&Bs[0][srow][scol + 8], brow + 8);
    wait_async0();
    __syncthreads();

    for (int s = 0; s < nsteps; ++s) {
        const int cur = s & 1, nxt = cur ^ 1;
        if (s + 1 < nsteps) {
            const _Float16* ap = arow + (size_t)(s + 1) * 32;
            const _Float16* bp = brow + (size_t)(s + 1) * 32;
            async_ld16(&As[nxt][srow][scol],     ap);
            async_ld16(&As[nxt][srow][scol + 8], ap + 8);
            async_ld16(&Bs[nxt][srow][scol],     bp);
            async_ld16(&Bs[nxt][srow][scol + 8], bp + 8);
        }
        v16h a = frag_ld(As[cur], wave * 16 + fr, kb);
        #pragma unroll
        for (int cg = 0; cg < 4; ++cg) {
            v16h bf = frag_ld(Bs[cur], cg * 16 + fr, kb);
            acc[cg] = wmma_f16(a, bf, acc[cg]);
        }
        wait_async0();
        __syncthreads();
    }

    const int hi8 = (lane >> 4) << 3;
    #pragma unroll
    for (int cg = 0; cg < 4; ++cg) {
        const int col = j0 + cg * 16 + fr;
        const float bb = bias ? bias[col] : 0.f;
        #pragma unroll
        for (int p = 0; p < 8; ++p) {
            const int row = i0 + wave * 16 + p + hi8;
            C[(size_t)row * ldc + col] = acc[cg][p] + bb;
        }
    }
}

// ---------------------------------------------------------------------------
// Fused big-tile kernel: per 64x64 output tile computes
//   w   = keys(i,:)·queries(j,:) / 512            (K = 128, WMMA)
//   m2  = pmT(i,:)·pmT(j,:) + EPS                 (K = 512, WMMA)
//   me  = exp(w) * m2
// mode 0: accumulate column sums c[j] += sum_i me  (LDS + global atomics)
// mode 1: out1[i,j] = me / (m2 * npi[j] + EPS)     (staged, coalesced store)
// Double-buffered async-to-LDS staging in both phases.
// ---------------------------------------------------------------------------
__global__ __launch_bounds__(128)
void ka_fused(const _Float16* __restrict__ Kh, const _Float16* __restrict__ Qh,
              const _Float16* __restrict__ PT, const float* __restrict__ npi,
              float* __restrict__ csum_out, float* __restrict__ out1, int mode) {
    __shared__ _Float16 As[2][64][40];
    __shared__ _Float16 Bs[2][64][40];
    __shared__ float Ot[64][65];
    __shared__ float csum[64];

    const int tid  = threadIdx.x;
    const int lane = tid & 31;
    const int wave = tid >> 5;
    const int fr   = lane & 15;
    const int kb   = (lane >> 4) << 3;
    const int hi8  = (lane >> 4) << 3;
    const int i0 = blockIdx.y * 64, j0 = blockIdx.x * 64;
    const int srow = tid >> 1, scol = (tid & 1) << 4;

    v8f acc1[4], acc2[4];
    #pragma unroll
    for (int c = 0; c < 4; ++c)
        #pragma unroll
        for (int e = 0; e < 8; ++e) { acc1[c][e] = 0.f; acc2[c][e] = 0.f; }

    // ---- Phase 1: keys x queries^T over D = 128 (4 k-steps) ----
    {
        const _Float16* arow = Kh + (size_t)(i0 + srow) * D_DIM + scol;
        const _Float16* brow = Qh + (size_t)(j0 + srow) * D_DIM + scol;
        async_ld16(&As[0][srow][scol],     arow);
        async_ld16(&As[0][srow][scol + 8], arow + 8);
        async_ld16(&Bs[0][srow][scol],     brow);
        async_ld16(&Bs[0][srow][scol + 8], brow + 8);
        wait_async0();
        __syncthreads();
        #pragma unroll
        for (int s = 0; s < (D_DIM >> 5); ++s) {
            const int cur = s & 1, nxt = cur ^ 1;
            if (s + 1 < (D_DIM >> 5)) {
                const _Float16* ap = arow + (size_t)(s + 1) * 32;
                const _Float16* bp = brow + (size_t)(s + 1) * 32;
                async_ld16(&As[nxt][srow][scol],     ap);
                async_ld16(&As[nxt][srow][scol + 8], ap + 8);
                async_ld16(&Bs[nxt][srow][scol],     bp);
                async_ld16(&Bs[nxt][srow][scol + 8], bp + 8);
            }
            v16h a = frag_ld(As[cur], wave * 16 + fr, kb);
            #pragma unroll
            for (int cg = 0; cg < 4; ++cg) {
                v16h bf = frag_ld(Bs[cur], cg * 16 + fr, kb);
                acc1[cg] = wmma_f16(a, bf, acc1[cg]);
            }
            wait_async0();
            __syncthreads();
        }
    }
    // ---- Phase 2: pmT Gram over B = 512 (16 k-steps) ----
    {
        const _Float16* arow = PT + (size_t)(i0 + srow) * B_BAGS + scol;
        const _Float16* brow = PT + (size_t)(j0 + srow) * B_BAGS + scol;
        async_ld16(&As[0][srow][scol],     arow);
        async_ld16(&As[0][srow][scol + 8], arow + 8);
        async_ld16(&Bs[0][srow][scol],     brow);
        async_ld16(&Bs[0][srow][scol + 8], brow + 8);
        wait_async0();
        __syncthreads();
        for (int s = 0; s < (B_BAGS >> 5); ++s) {
            const int cur = s & 1, nxt = cur ^ 1;
            if (s + 1 < (B_BAGS >> 5)) {
                const _Float16* ap = arow + (size_t)(s + 1) * 32;
                const _Float16* bp = brow + (size_t)(s + 1) * 32;
                async_ld16(&As[nxt][srow][scol],     ap);
                async_ld16(&As[nxt][srow][scol + 8], ap + 8);
                async_ld16(&Bs[nxt][srow][scol],     bp);
                async_ld16(&Bs[nxt][srow][scol + 8], bp + 8);
            }
            v16h a = frag_ld(As[cur], wave * 16 + fr, kb);
            #pragma unroll
            for (int cg = 0; cg < 4; ++cg) {
                v16h bf = frag_ld(Bs[cur], cg * 16 + fr, kb);
                acc2[cg] = wmma_f16(a, bf, acc2[cg]);
            }
            wait_async0();
            __syncthreads();
        }
    }

    if (mode == 0) {
        if (tid < 64) csum[tid] = 0.f;
        __syncthreads();
    }

    #pragma unroll
    for (int cg = 0; cg < 4; ++cg) {
        float npiv = 0.f;
        if (mode == 1) npiv = npi[j0 + cg * 16 + fr];
        float s = 0.f;
        #pragma unroll
        for (int p = 0; p < 8; ++p) {
            const float w   = acc1[cg][p] * INV_BATCH;
            const float m2v = acc2[cg][p] + EPS;
            const float me  = __expf(w) * m2v;
            if (mode == 0) {
                s += me;
            } else {
                const float o = me / (m2v * npiv + EPS);
                Ot[wave * 16 + p + hi8][cg * 16 + fr] = o;
            }
        }
        if (mode == 0) atomicAdd(&csum[cg * 16 + fr], s);
    }
    __syncthreads();
    if (mode == 0) {
        if (tid < 64) atomicAdd(&csum_out[j0 + tid], csum[tid]);
    } else {
        for (int idx = tid; idx < 64 * 64; idx += 128) {
            const int r = idx >> 6, cc = idx & 63;
            out1[(size_t)(i0 + r) * N_INST + (j0 + cc)] = Ot[r][cc];
        }
    }
}

// ------------------------- small helper kernels ----------------------------
__global__ void ka_cvt(const float* __restrict__ s, _Float16* __restrict__ d, int n) {
    const int i = blockIdx.x * 256 + threadIdx.x;
    if (i < n) d[i] = (_Float16)s[i];
}

__global__ void ka_zero(float* __restrict__ p, int n) {
    const int i = blockIdx.x * 256 + threadIdx.x;
    if (i < n) p[i] = 0.f;
}

// pmT[n][b] = f16(mask[b][n] + EPS); coalesced reads of mask.
__global__ void ka_pmt(const float* __restrict__ mask, _Float16* __restrict__ pmT) {
    const int i = blockIdx.x * 256 + threadIdx.x;  // over B*N
    if (i >= B_BAGS * N_INST) return;
    const int b = i / N_INST, n = i - b * N_INST;
    pmT[(size_t)n * B_BAGS + b] = (_Float16)(mask[i] + EPS);
}

// Per-feature biased batch stats over the instance dim. grid = D blocks.
__global__ __launch_bounds__(256)
void ka_bn_stats(const float* __restrict__ raw, float* __restrict__ mean,
                 float* __restrict__ rstd) {
    __shared__ float ss[256], s2[256];
    const int d = blockIdx.x;
    float a = 0.f, b = 0.f;
    for (int n = threadIdx.x; n < N_INST; n += 256) {
        const float x = raw[(size_t)n * D_DIM + d];
        a += x; b += x * x;
    }
    ss[threadIdx.x] = a; s2[threadIdx.x] = b;
    __syncthreads();
    for (int t = 128; t > 0; t >>= 1) {
        if (threadIdx.x < t) {
            ss[threadIdx.x] += ss[threadIdx.x + t];
            s2[threadIdx.x] += s2[threadIdx.x + t];
        }
        __syncthreads();
    }
    if (threadIdx.x == 0) {
        const float m = ss[0] * (1.0f / N_INST);
        const float v = s2[0] * (1.0f / N_INST) - m * m;
        mean[d] = m;
        rstd[d] = rsqrtf(v + BN_EPS);
    }
}

__global__ void ka_bn_apply(const float* __restrict__ raw, const float* __restrict__ mean,
                            const float* __restrict__ rstd, const float* __restrict__ g,
                            const float* __restrict__ beta, _Float16* __restrict__ oh,
                            float* __restrict__ of) {
    const int i = blockIdx.x * 256 + threadIdx.x;  // N*D, exact multiple
    const int d = i & (D_DIM - 1);
    const float y = (raw[i] - mean[d]) * rstd[d] * g[d] + beta[d];
    oh[i] = (_Float16)y;
    if (of) of[i] = y;
}

// bag_norm[b] = sum_n c[n] * (mask[b][n] + EPS); grid = B blocks.
__global__ __launch_bounds__(256)
void ka_bagnorm(const float* __restrict__ c, const float* __restrict__ mask,
                float* __restrict__ bagn) {
    __shared__ float red[256];
    const int b = blockIdx.x;
    float s = 0.f;
    for (int n = threadIdx.x; n < N_INST; n += 256)
        s += c[n] * (mask[(size_t)b * N_INST + n] + EPS);
    red[threadIdx.x] = s;
    __syncthreads();
    for (int t = 128; t > 0; t >>= 1) {
        if (threadIdx.x < t) red[threadIdx.x] += red[threadIdx.x + t];
        __syncthreads();
    }
    if (threadIdx.x == 0) bagn[b] = red[0];
}

// npi[n] = sum_b bag_norm[b]*(mask[b][n]+EPS);  pw[n] = c[n]/(EPS+npi[n])
__global__ __launch_bounds__(256)
void ka_npi(const float* __restrict__ bagn, const float* __restrict__ mask,
            const float* __restrict__ c, float* __restrict__ npi,
            float* __restrict__ pw) {
    __shared__ float bn_s[B_BAGS];
    for (int t = threadIdx.x; t < B_BAGS; t += 256) bn_s[t] = bagn[t];
    __syncthreads();
    const int n = blockIdx.x * 256 + threadIdx.x;  // exact N
    float s = 0.f;
    #pragma unroll 4
    for (int b = 0; b < B_BAGS; ++b)
        s += bn_s[b] * (mask[(size_t)b * N_INST + n] + EPS);
    npi[n] = s;
    pw[n] = c[n] / (EPS + s);
}

// out2[b][d] = sum_n (mask[b][n]+EPS) * pw[n] * keys_bn[n][d]; grid = B, block = D.
__global__ __launch_bounds__(128)
void ka_bagw(const float* __restrict__ mask, const float* __restrict__ pw,
             const float* __restrict__ kbn, float* __restrict__ out2) {
    const int b = blockIdx.x, d = threadIdx.x;
    float s = 0.f;
    #pragma unroll 4
    for (int n = 0; n < N_INST; ++n) {
        const float w = (mask[(size_t)b * N_INST + n] + EPS) * pw[n];
        s += w * kbn[(size_t)n * D_DIM + d];
    }
    out2[(size_t)b * D_DIM + d] = s;
}

// ---------------------------------------------------------------------------
extern "C" void kernel_launch(void* const* d_in, const int* in_sizes, int n_in,
                              void* d_out, int out_size, void* d_ws, size_t ws_size,
                              hipStream_t stream) {
    const float* xprot = (const float*)d_in[0];
    const float* mask  = (const float*)d_in[1];
    const float* Wk    = (const float*)d_in[2];
    const float* bk    = (const float*)d_in[3];
    const float* gk    = (const float*)d_in[4];
    const float* betak = (const float*)d_in[5];
    const float* Wq    = (const float*)d_in[6];
    const float* bq    = (const float*)d_in[7];
    const float* gq    = (const float*)d_in[8];
    const float* betaq = (const float*)d_in[9];

    char* ws = (char*)d_ws;
    size_t off = 0;
    auto carve = [&](size_t bytes) -> void* {
        off = (off + 255) & ~(size_t)255;
        void* p = ws + off;
        off += bytes;
        return p;
    };
    // total workspace ~32.1 MB
    _Float16* xprot_h = (_Float16*)carve((size_t)N_INST * P_DIM * 2);
    _Float16* wk_h    = (_Float16*)carve((size_t)D_DIM * P_DIM * 2);
    _Float16* wq_h    = (_Float16*)carve((size_t)D_DIM * P_DIM * 2);
    float*    kraw    = (float*)carve((size_t)N_INST * D_DIM * 4);
    float*    qraw    = (float*)carve((size_t)N_INST * D_DIM * 4);
    float*    kbn     = (float*)carve((size_t)N_INST * D_DIM * 4);
    _Float16* kh      = (_Float16*)carve((size_t)N_INST * D_DIM * 2);
    _Float16* qh      = (_Float16*)carve((size_t)N_INST * D_DIM * 2);
    _Float16* pmT     = (_Float16*)carve((size_t)N_INST * B_BAGS * 2);
    float*    meanK   = (float*)carve(D_DIM * 4);
    float*    rstdK   = (float*)carve(D_DIM * 4);
    float*    meanQ   = (float*)carve(D_DIM * 4);
    float*    rstdQ   = (float*)carve(D_DIM * 4);
    float*    cvec    = (float*)carve(N_INST * 4);
    float*    bagn    = (float*)carve(B_BAGS * 4);
    float*    npi     = (float*)carve(N_INST * 4);
    float*    pw      = (float*)carve(N_INST * 4);

    float* out1 = (float*)d_out;
    float* out2 = out1 + (size_t)N_INST * N_INST;

    // 1) convert inputs to f16
    ka_cvt<<<(N_INST * P_DIM) / 256, 256, 0, stream>>>(xprot, xprot_h, N_INST * P_DIM);
    ka_cvt<<<(D_DIM * P_DIM) / 256, 256, 0, stream>>>(Wk, wk_h, D_DIM * P_DIM);
    ka_cvt<<<(D_DIM * P_DIM) / 256, 256, 0, stream>>>(Wq, wq_h, D_DIM * P_DIM);

    // 2) keysraw = xprot x Wk^T + bk ; queriesraw = xprot x Wq^T + bq (WMMA)
    {
        dim3 g(D_DIM / 64, N_INST / 64);
        ka_gemm_nt<<<g, 128, 0, stream>>>(xprot_h, wk_h, bk, kraw, P_DIM, P_DIM, P_DIM, D_DIM);
        ka_gemm_nt<<<g, 128, 0, stream>>>(xprot_h, wq_h, bq, qraw, P_DIM, P_DIM, P_DIM, D_DIM);
    }

    // 3) batch-norm (training-mode biased stats over instances)
    ka_bn_stats<<<D_DIM, 256, 0, stream>>>(kraw, meanK, rstdK);
    ka_bn_stats<<<D_DIM, 256, 0, stream>>>(qraw, meanQ, rstdQ);
    ka_bn_apply<<<(N_INST * D_DIM) / 256, 256, 0, stream>>>(kraw, meanK, rstdK, gk, betak, kh, kbn);
    ka_bn_apply<<<(N_INST * D_DIM) / 256, 256, 0, stream>>>(qraw, meanQ, rstdQ, gq, betaq, qh, (float*)nullptr);

    // 4) pmT (N,B) in f16; pm = mask + EPS
    ka_pmt<<<(B_BAGS * N_INST) / 256, 256, 0, stream>>>(mask, pmT);

    // 5) pass 0 of fused tile kernel: column sums c[j] of masked_exps
    ka_zero<<<(N_INST + 255) / 256, 256, 0, stream>>>(cvec, N_INST);
    {
        dim3 g(N_INST / 64, N_INST / 64);
        ka_fused<<<g, 128, 0, stream>>>(kh, qh, pmT, npi, cvec, out1, 0);
    }

    // 6) bag_norm = pm @ c ; npi = bag_norm @ pm ; pw = c / (EPS + npi)
    ka_bagnorm<<<B_BAGS, 256, 0, stream>>>(cvec, mask, bagn);
    ka_npi<<<N_INST / 256, 256, 0, stream>>>(bagn, mask, cvec, npi, pw);

    // 7) pass 1: out1 = masked_exps / (m2 * npi[j] + EPS)
    {
        dim3 g(N_INST / 64, N_INST / 64);
        ka_fused<<<g, 128, 0, stream>>>(kh, qh, pmT, npi, cvec, out1, 1);
    }

    // 8) bag_weighted = pm @ (values * pw)
    ka_bagw<<<B_BAGS, 128, 0, stream>>>(mask, pw, kbn, out2);
}
